// DFADNANet_7876970020897
// MI455X (gfx1250) — matrix operand order repaired and checked by reading, not compiled, not run on hardware
//
#include <hip/hip_runtime.h>
#include <hip/hip_bf16.h>

// ---------------------------------------------------------------------------
// Types for CDNA5 WMMA
// ---------------------------------------------------------------------------
typedef __attribute__((ext_vector_type(16))) __bf16        v16bf;
typedef __attribute__((ext_vector_type(8)))  float         v8f;
typedef __attribute__((ext_vector_type(4)))  unsigned int  v4u;

#define N_NODES 50000
#define N_EDGES 300000
#define IN_CH   128
#define HIDDEN  64
#define OUT_CH  40
#define OUT_PAD 48
#define NLAYERS 4
#define HEADS   8
#define HEAD_D  8

// ---------------------------------------------------------------------------
// bf16 helpers (round-to-nearest-even), stored as unsigned short
// ---------------------------------------------------------------------------
__device__ __forceinline__ unsigned short f2bf(float f) {
    unsigned int u = __float_as_uint(f);
    u += 0x7fffu + ((u >> 16) & 1u);
    return (unsigned short)(u >> 16);
}
__device__ __forceinline__ float bf2f(unsigned short h) {
    return __uint_as_float(((unsigned int)h) << 16);
}

// ---------------------------------------------------------------------------
// Small utility kernels
// ---------------------------------------------------------------------------
__global__ void fill_f32(float* p, float v, int n) {
    int i = blockIdx.x * blockDim.x + threadIdx.x;
    if (i < n) p[i] = v;
}

__global__ void cvt_f32_to_bf16(const float* __restrict__ s, unsigned short* __restrict__ d, int n) {
    int i = blockIdx.x * blockDim.x + threadIdx.x;
    if (i < n) d[i] = f2bf(s[i]);
}

__global__ void relu_cvt_bf16(const float* __restrict__ s, unsigned short* __restrict__ d, int n) {
    int i = blockIdx.x * blockDim.x + threadIdx.x;
    if (i < n) d[i] = f2bf(fmaxf(s[i], 0.0f));
}

// deg[dst] += 1 per edge (deg pre-filled with 1.0 for the self loop)
__global__ void deg_accum(const long long* __restrict__ ei, float* __restrict__ deg, int E) {
    int e = blockIdx.x * blockDim.x + threadIdx.x;
    if (e < E) atomicAdd(&deg[(int)ei[E + e]], 1.0f);
}

__global__ void rsqrt_ip(float* p, int n) {
    int i = blockIdx.x * blockDim.x + threadIdx.x;
    if (i < n) { float d = p[i]; p[i] = (d > 0.0f) ? rsqrtf(d) : 0.0f; }
}

__global__ void edge_norm(const long long* __restrict__ ei, const float* __restrict__ dinv,
                          float* __restrict__ norm, int E) {
    int e = blockIdx.x * blockDim.x + threadIdx.x;
    if (e < E) norm[e] = dinv[(int)ei[e]] * dinv[(int)ei[E + e]];
}

// ---------------------------------------------------------------------------
// Weight pack: fp32 row-major W (K x Nc) -> bf16 fragment-ready layout
//   dst[((ks*(Nn/16) + tn)*32 + lane)*16 + t]
//     = W[(ks*32 + (lane>=16?8:0) + (t&7) + (t>>3)*16) * Nc + tn*16 + (lane&15)]   (0 if col>=Nc)
// This is exactly the per-lane B fragment of v_wmma_f32_16x16x32_bf16, so the
// GEMM reads each fragment as two contiguous b128 loads (no scalar gathers).
// ---------------------------------------------------------------------------
__global__ void pack_weight_frag(const float* __restrict__ W, unsigned short* __restrict__ dst,
                                 int K, int Nc, int Nn)
{
    const int idx   = blockIdx.x * blockDim.x + threadIdx.x;
    const int total = (K >> 5) * (Nn >> 4) * 512;
    if (idx >= total) return;
    const int t     = idx & 15;
    const int lane  = (idx >> 4) & 31;
    const int tile  = idx >> 9;
    const int tn    = tile % (Nn >> 4);
    const int ks    = tile / (Nn >> 4);
    const int col   = tn * 16 + (lane & 15);
    const int kr    = ks * 32 + (lane >> 4) * 8 + (t & 7) + (t >> 3) * 16;
    const float val = (col < Nc) ? W[(size_t)kr * Nc + col] : 0.0f;
    dst[idx] = f2bf(val);
}

// ---------------------------------------------------------------------------
// WMMA bf16 GEMM:  C[M x (NT*16)] = act(A[M x (KSTEPS*32)] @ B + bias)
// One wave per 16-row strip; the A fragment is loaded once per K-step and
// reused across NT column tiles (NT accumulators). KSTEPS/NT are template
// params so the MAC loop fully unrolls: 2 b128 A-loads + 2*NT b128 B-loads
// + NT v_wmma per K-step, branch-free. Requires M % 16 == 0 (true here).
//
// A fragment per ISA 7.12.2 (16-bit A 16x32): lanes 0-15 row=lane&15, K in
// {kk..kk+7, kk+16..kk+23}; lanes 16-31 same row, K shifted by +8.
// C/D: lanes 0-15 -> rows 0-7 (vgpr r), lanes 16-31 -> rows 8-15.
// ---------------------------------------------------------------------------
template <int KSTEPS, int NT>
__global__ void wmma_gemm_bf16(const unsigned short* __restrict__ A, int lda,
                               const unsigned short* __restrict__ Bp,
                               const float* __restrict__ bias, int biasN,
                               float* __restrict__ outF, unsigned short* __restrict__ outB,
                               int ldc, int M, int relu)
{
    const int lane   = threadIdx.x & 31;
    const int wave   = threadIdx.x >> 5;
    const int strips = M >> 4;
    const int strip  = blockIdx.x * (blockDim.x >> 5) + wave;
    if (strip >= strips) return;            // wave-uniform: EXEC stays all-ones

    const int hi16 = lane >> 4;             // lane group 0 / 1
    const int l15  = lane & 15;

    const unsigned short* Arow = A + (size_t)(strip * 16 + l15) * lda;

    v8f c[NT];
    #pragma unroll
    for (int t = 0; t < NT; ++t) c[t] = (v8f){0.f,0.f,0.f,0.f,0.f,0.f,0.f,0.f};

    #pragma unroll
    for (int ks = 0; ks < KSTEPS; ++ks) {
        // ---- A fragment: two 16B vector loads, reused by all NT WMMAs
        union { v16bf v; v4u q[2]; } a;
        const int k0 = ks * 32 + hi16 * 8;
        a.q[0] = *(const v4u*)(Arow + k0);        // K = k0 .. k0+7
        a.q[1] = *(const v4u*)(Arow + k0 + 16);   // K = k0+16 .. k0+23
        #pragma unroll
        for (int t = 0; t < NT; ++t) {
            // ---- B fragment: packed, two contiguous 16B loads
            union { v16bf v; v4u q[2]; } b;
            const unsigned short* bp = Bp + (((size_t)ks * NT + t) * 32 + lane) * 16;
            b.q[0] = *(const v4u*)(bp);
            b.q[1] = *(const v4u*)(bp + 8);
            c[t] = __builtin_amdgcn_wmma_f32_16x16x32_bf16(
                       false, a.v, false, b.v, (short)0, c[t], false, false);
        }
    }

    const int rbase = strip * 16 + hi16 * 8;
    #pragma unroll
    for (int t = 0; t < NT; ++t) {
        const int   bcol = t * 16 + l15;
        const float bv   = (bias != nullptr && bcol < biasN) ? bias[bcol] : 0.0f;
        #pragma unroll
        for (int r = 0; r < 8; ++r) {
            float val = c[t][r] + bv;
            if (relu) val = fmaxf(val, 0.0f);
            const size_t o = (size_t)(rbase + r) * ldc + bcol;
            if (outF) outF[o] = val;
            if (outB) outB[o] = f2bf(val);
        }
    }
}

// ---------------------------------------------------------------------------
// Edge attention + norm-scaled segment sum.
// One thread per (edge, head). Self loops are indices e >= E (node n = e-E).
// q/k/v are bf16, each head-row is a 16B-aligned octet -> single b128 gathers.
// k/v layout: (layer l, node n, 64) contiguous == GEMM row l*N + n.
// ---------------------------------------------------------------------------
__global__ void attn_edge(const long long* __restrict__ ei,
                          const float* __restrict__ dinv,
                          const float* __restrict__ norm,
                          const unsigned short* __restrict__ q,
                          const unsigned short* __restrict__ k,
                          const unsigned short* __restrict__ v,
                          float* __restrict__ agg, int L, int E, int Nn)
{
    const int idx   = blockIdx.x * blockDim.x + threadIdx.x;
    const int total = (E + Nn) * HEADS;
    if (idx >= total) return;
    const int e  = idx >> 3;
    const int hd = idx & 7;

    int s, d; float w;
    if (e < E) {
        s = (int)ei[e]; d = (int)ei[E + e]; w = norm[e];
    } else {
        const int n = e - E; s = n; d = n;
        const float di = dinv[n]; w = di * di;
    }

    const int off = hd * HEAD_D;
    union Oct { v4u q; unsigned short u[8]; };

    Oct qo; qo.q = *(const v4u*)(q + ((size_t)d * HIDDEN + off));
    float qv[HEAD_D];
    #pragma unroll
    for (int j = 0; j < HEAD_D; ++j) qv[j] = bf2f(qo.u[j]);

    const float scale = 0.35355339059327373f;   // 1/sqrt(8)
    float sc[NLAYERS];
    float m = -1e30f;
    for (int l = 0; l < L; ++l) {
        Oct ko; ko.q = *(const v4u*)(k + (((size_t)l * Nn + s) * HIDDEN + off));
        float acc = 0.0f;
        #pragma unroll
        for (int j = 0; j < HEAD_D; ++j) acc += qv[j] * bf2f(ko.u[j]);
        sc[l] = acc * scale;
        m = fmaxf(m, sc[l]);
    }
    float se = 0.0f;
    for (int l = 0; l < L; ++l) { sc[l] = __expf(sc[l] - m); se += sc[l]; }
    const float winv = w / se;

    float msg[HEAD_D] = {0.f, 0.f, 0.f, 0.f, 0.f, 0.f, 0.f, 0.f};
    for (int l = 0; l < L; ++l) {
        Oct vo; vo.q = *(const v4u*)(v + (((size_t)l * Nn + s) * HIDDEN + off));
        const float a = sc[l];
        #pragma unroll
        for (int j = 0; j < HEAD_D; ++j) msg[j] += a * bf2f(vo.u[j]);
    }

    float* ap = agg + ((size_t)d * HIDDEN + off);
    #pragma unroll
    for (int j = 0; j < HEAD_D; ++j) atomicAdd(ap + j, msg[j] * winv);
}

// ---------------------------------------------------------------------------
// Per-node log-softmax over 40 logits (ld = 48)
// ---------------------------------------------------------------------------
__global__ void log_softmax_k(const float* __restrict__ logits, float* __restrict__ out, int Nn)
{
    const int n = blockIdx.x * blockDim.x + threadIdx.x;
    if (n >= Nn) return;
    const float* lp = logits + (size_t)n * OUT_PAD;
    float m = -1e30f;
    for (int c = 0; c < OUT_CH; ++c) m = fmaxf(m, lp[c]);
    float se = 0.0f;
    for (int c = 0; c < OUT_CH; ++c) se += __expf(lp[c] - m);
    const float ls = m + __logf(se);
    float* op = out + (size_t)n * OUT_CH;
    for (int c = 0; c < OUT_CH; ++c) op[c] = lp[c] - ls;
}

// ---------------------------------------------------------------------------
// Host orchestration
// ---------------------------------------------------------------------------
extern "C" void kernel_launch(void* const* d_in, const int* in_sizes, int n_in,
                              void* d_out, int out_size, void* d_ws, size_t ws_size,
                              hipStream_t stream)
{
    (void)in_sizes; (void)n_in; (void)out_size; (void)ws_size;

    const float*     x  = (const float*)d_in[0];
    const long long* ei = (const long long*)d_in[1];   // int64 (2, E)
    const float*     w1 = (const float*)d_in[3];
    const float*     b1 = (const float*)d_in[4];
    const float*     wq = (const float*)d_in[5];
    const float*     wk = (const float*)d_in[6];
    const float*     wv = (const float*)d_in[7];
    const float*     w2 = (const float*)d_in[8];
    const float*     b2 = (const float*)d_in[9];
    float* out = (float*)d_out;

    const int N = N_NODES, E = N_EDGES, C = HIDDEN;

    // ---- workspace carve-out (256B aligned)
    char*  ws  = (char*)d_ws;
    size_t off = 0;
    auto alloc = [&](size_t bytes) -> char* {
        char* p = ws + off;
        off = (off + bytes + 255) & ~(size_t)255;
        return p;
    };
    unsigned short* Xbf  = (unsigned short*)alloc((size_t)N * IN_CH * 2);
    unsigned short* w1p  = (unsigned short*)alloc((size_t)IN_CH * C * 2);           // packed
    unsigned short* wqp  = (unsigned short*)alloc((size_t)NLAYERS * C * C * 2);     // packed
    unsigned short* wkp  = (unsigned short*)alloc((size_t)NLAYERS * C * C * 2);     // packed
    unsigned short* wvp  = (unsigned short*)alloc((size_t)NLAYERS * C * C * 2);     // packed
    unsigned short* w2p  = (unsigned short*)alloc((size_t)C * OUT_PAD * 2);         // packed
    unsigned short* xall = (unsigned short*)alloc((size_t)(NLAYERS + 1) * N * C * 2); // layer-major
    unsigned short* qbf  = (unsigned short*)alloc((size_t)N * C * 2);
    unsigned short* kbf  = (unsigned short*)alloc((size_t)NLAYERS * N * C * 2);
    unsigned short* vbf  = (unsigned short*)alloc((size_t)NLAYERS * N * C * 2);
    float* deg    = (float*)alloc((size_t)N * 4);             // becomes dinv in place
    float* norm   = (float*)alloc((size_t)E * 4);
    float* agg    = (float*)alloc((size_t)N * C * 4);
    float* logits = (float*)alloc((size_t)N * OUT_PAD * 4);

    const int BT = 256;
    auto nb = [&](int n) { return (n + BT - 1) / BT; };

    // ---- input conversion + fragment-packing of all weights (fuses bf16 cvt)
    cvt_f32_to_bf16<<<nb(N * IN_CH), BT, 0, stream>>>(x, Xbf, N * IN_CH);
    pack_weight_frag<<<nb(IN_CH * C), BT, 0, stream>>>(w1, w1p, IN_CH, C, C);
    for (int i = 0; i < NLAYERS; ++i) {
        pack_weight_frag<<<nb(C * C), BT, 0, stream>>>(wq + i * C * C, wqp + i * C * C, C, C, C);
        pack_weight_frag<<<nb(C * C), BT, 0, stream>>>(wk + i * C * C, wkp + i * C * C, C, C, C);
        pack_weight_frag<<<nb(C * C), BT, 0, stream>>>(wv + i * C * C, wvp + i * C * C, C, C, C);
    }
    pack_weight_frag<<<nb(C * OUT_PAD), BT, 0, stream>>>(w2, w2p, C, OUT_CH, OUT_PAD);

    // ---- degree / symmetric norm (self loop contributes the initial 1.0)
    fill_f32<<<nb(N), BT, 0, stream>>>(deg, 1.0f, N);
    deg_accum<<<nb(E), BT, 0, stream>>>(ei, deg, E);
    rsqrt_ip<<<nb(N), BT, 0, stream>>>(deg, N);               // deg -> dinv
    edge_norm<<<nb(E), BT, 0, stream>>>(ei, deg, norm, E);

    auto gblk = [](int M) { return (M / 16 + 3) / 4; };       // 4 waves (strips) / block

    // ---- h = relu(x @ w1 + b1)  ->  x_all layer 0 (bf16)   [K=128 -> 4 ksteps]
    wmma_gemm_bf16<4, 4><<<gblk(N), 128, 0, stream>>>(
        Xbf, IN_CH, w1p, b1, C, nullptr, xall, C, N, 1);

    // ---- DNAConv layers
    for (int i = 0; i < NLAYERS; ++i) {
        const int L = i + 1;
        // q from the newest layer slice
        wmma_gemm_bf16<2, 4><<<gblk(N), 128, 0, stream>>>(
            xall + (size_t)(L - 1) * N * C, C, wqp + i * C * C,
            nullptr, 0, nullptr, qbf, C, N, 0);
        // k, v across all L layer slices in a single GEMM (layer-major => contiguous)
        wmma_gemm_bf16<2, 4><<<gblk(N * L), 128, 0, stream>>>(
            xall, C, wkp + i * C * C, nullptr, 0, nullptr, kbf, C, N * L, 0);
        wmma_gemm_bf16<2, 4><<<gblk(N * L), 128, 0, stream>>>(
            xall, C, wvp + i * C * C, nullptr, 0, nullptr, vbf, C, N * L, 0);

        hipMemsetAsync(agg, 0, (size_t)N * C * sizeof(float), stream);
        const int tot = (E + N) * HEADS;
        attn_edge<<<nb(tot), BT, 0, stream>>>(ei, deg, norm, qbf, kbf, vbf, agg, L, E, N);

        // relu + append to x_all (bf16)
        relu_cvt_bf16<<<nb(N * C), BT, 0, stream>>>(agg, xall + (size_t)L * N * C, N * C);
    }

    // ---- logits = x_all[:, -1] @ w2 + b2 ; then log_softmax  [K=64, Nn=48]
    wmma_gemm_bf16<2, 3><<<gblk(N), 128, 0, stream>>>(
        xall + (size_t)NLAYERS * N * C, C, w2p, b2, OUT_CH, logits, nullptr, OUT_PAD, N, 0);
    log_softmax_k<<<nb(N), BT, 0, stream>>>(logits, out, N);
}
